// MulitLatentAttention_90305982365814
// MI455X (gfx1250) — compile-verified, matrix operand order checked by reading, not run on hardware
//
#include <hip/hip_runtime.h>
#include <hip/hip_bf16.h>

// ---------------------------------------------------------------------------
// Types for CDNA5 WMMA (wave32).
// ---------------------------------------------------------------------------
typedef __bf16          bf16_t;
typedef bf16_t          bf16x16 __attribute__((ext_vector_type(16)));
typedef bf16_t          bf16x2  __attribute__((ext_vector_type(2)));
typedef float           f32x8   __attribute__((ext_vector_type(8)));
typedef unsigned short  u16;
typedef u16             u16x8   __attribute__((ext_vector_type(8)));
typedef u16             u16x16  __attribute__((ext_vector_type(16)));
typedef unsigned        u32x8   __attribute__((ext_vector_type(8)));

// Problem constants (match reference).
#define MLA_S    2048
#define MLA_HID  2048
#define MLA_H    16
#define MLA_DN   128
#define MLA_DR   64
#define MLA_DV   128
#define MLA_QLR  1536
#define MLA_KVLR 512
#define MLA_DQK  192   // DN + DR

// ---------------------------------------------------------------------------
// fp32 -> bf16 conversion (lowers to v_cvt_pk_bf16_f32 on gfx1250).
// ---------------------------------------------------------------------------
__device__ __forceinline__ unsigned pk2bf(float a, float b) {
#if __has_builtin(__builtin_amdgcn_cvt_pk_bf16_f32)
  return __builtin_bit_cast(unsigned, __builtin_amdgcn_cvt_pk_bf16_f32(a, b));
#else
  bf16x2 p;
  p[0] = (bf16_t)a;
  p[1] = (bf16_t)b;
  return __builtin_bit_cast(unsigned, p);
#endif
}
__device__ __forceinline__ u16 f2bf(float f) {
  bf16_t h = (bf16_t)f;
  return __builtin_bit_cast(u16, h);
}

// 16-element bf16 A/B fragment (16-bit WMMA layout) from an LDS row.
// p = row base + k-offset (already includes +8 for lanes 16..31).
// Elements 0..7 = k..k+7, elements 8..15 = k+16..k+23.
__device__ __forceinline__ bf16x16 frag_from_lds(const u16* p) {
  u16x8 lo = *(const u16x8*)(p);
  u16x8 hi = *(const u16x8*)(p + 16);
  u16x16 f;
#pragma unroll
  for (int i = 0; i < 8; ++i) { f[i] = lo[i]; f[i + 8] = hi[i]; }
  return __builtin_bit_cast(bf16x16, f);
}

// Same fragment mapping sourced from a fp32 row in global, converting
// in-flight with packed cvt. kstart already includes (lane>>4)*8.
__device__ __forceinline__ bf16x16 frag_from_global(const float* row, int kstart) {
  const float4 a = *(const float4*)(row + kstart);
  const float4 b = *(const float4*)(row + kstart + 4);
  const float4 c = *(const float4*)(row + kstart + 16);
  const float4 d = *(const float4*)(row + kstart + 20);
  u32x8 f;
  f[0] = pk2bf(a.x, a.y); f[1] = pk2bf(a.z, a.w);
  f[2] = pk2bf(b.x, b.y); f[3] = pk2bf(b.z, b.w);
  f[4] = pk2bf(c.x, c.y); f[5] = pk2bf(c.z, c.w);
  f[6] = pk2bf(d.x, d.y); f[7] = pk2bf(d.z, d.w);
  return __builtin_bit_cast(bf16x16, f);
}

__device__ __forceinline__ float hmax16(float v) {
  v = fmaxf(v, __shfl_xor(v, 1, 32));
  v = fmaxf(v, __shfl_xor(v, 2, 32));
  v = fmaxf(v, __shfl_xor(v, 4, 32));
  v = fmaxf(v, __shfl_xor(v, 8, 32));
  return v;
}
__device__ __forceinline__ float hsum16(float v) {
  v += __shfl_xor(v, 1, 32);
  v += __shfl_xor(v, 2, 32);
  v += __shfl_xor(v, 4, 32);
  v += __shfl_xor(v, 8, 32);
  return v;
}

// ---------------------------------------------------------------------------
// Generic GEMM: C[M,N] = A[M,K] * B[K,N], fp32 in/out, bf16 WMMA math.
// Block = 256 threads (8 waves). Block tile 128x128, K stage 64.
// Wave tile 32x64 (2 M-frags x 4 N-frags -> 8 f32x8 accumulators).
// Staging issues all global loads before converting, so the 8 outstanding
// global_load_b128s overlap (drained with descending s_wait_loadcnt).
// M, K must be multiples of 128/64 (true for all uses); N may be ragged.
// ---------------------------------------------------------------------------
#define GBM 128
#define GBN 128
#define GBK 64
#define GLDK (GBK + 8)   // u16 stride, 144B rows: 16B aligned, conflict pad

__global__ __launch_bounds__(256) void gemm_bf16_wmma(
    const float* __restrict__ A, const float* __restrict__ B,
    float* __restrict__ C, int M, int N, int K) {
  __shared__ u16 sA[GBM][GLDK];
  __shared__ u16 sB[GBN][GLDK];

  const int tid  = threadIdx.x;
  const int lane = tid & 31;
  const int wave = tid >> 5;
  const int wr   = wave & 3;   // 4 waves along M (4 * 32 = 128)
  const int wc   = wave >> 2;  // 2 waves along N (2 * 64 = 128)
  const int hb   = lane >> 4;  // half-wave id (k-mapping offset)
  const int ln   = lane & 15;
  const int m0   = blockIdx.y * GBM;
  const int n0   = blockIdx.x * GBN;

  f32x8 acc[2][4];
#pragma unroll
  for (int i = 0; i < 2; ++i)
#pragma unroll
    for (int j = 0; j < 4; ++j) acc[i][j] = (f32x8)0.0f;

  const bool bFast = (n0 + ((tid & 31) * 4) + 3) < N;  // per-thread B fast path

  for (int kb = 0; kb < K; kb += GBK) {
    __syncthreads();
    // ---- stage A (128 x 64): issue 8 loads, then convert+store ----
    {
      const int c4 = tid & 15;
      const int r0 = tid >> 4;
      float4 va[8];
#pragma unroll
      for (int p = 0; p < 8; ++p)
        va[p] = *(const float4*)(A + (size_t)(m0 + r0 + p * 16) * K + kb + c4 * 4);
#pragma unroll
      for (int p = 0; p < 8; ++p) {
        const unsigned lo = pk2bf(va[p].x, va[p].y);
        const unsigned hi = pk2bf(va[p].z, va[p].w);
        *(unsigned long long*)(&sA[r0 + p * 16][c4 * 4]) =
            (unsigned long long)lo | ((unsigned long long)hi << 32);
      }
    }
    // ---- stage B (64 x 128) transposed -> sB[n][k] ----
    {
      const int c4 = tid & 31;
      const int k0 = tid >> 5;
      const int n  = c4 * 4;
      float4 vb[8];
      if (bFast) {
#pragma unroll
        for (int p = 0; p < 8; ++p)
          vb[p] = *(const float4*)(B + (size_t)(kb + k0 + p * 8) * N + n0 + n);
      } else {
#pragma unroll
        for (int p = 0; p < 8; ++p) {
          float t[4];
#pragma unroll
          for (int j = 0; j < 4; ++j) {
            const int nn = n0 + n + j;
            t[j] = (nn < N) ? B[(size_t)(kb + k0 + p * 8) * N + nn] : 0.0f;
          }
          vb[p] = make_float4(t[0], t[1], t[2], t[3]);
        }
      }
#pragma unroll
      for (int p = 0; p < 8; ++p) {
        const int k = k0 + p * 8;
        sB[n + 0][k] = f2bf(vb[p].x);
        sB[n + 1][k] = f2bf(vb[p].y);
        sB[n + 2][k] = f2bf(vb[p].z);
        sB[n + 3][k] = f2bf(vb[p].w);
      }
    }
    __syncthreads();

    // Prefetch next K stage into cache (global_prefetch_b8 on gfx1250).
    if (kb + GBK < K) {
      __builtin_prefetch(A + (size_t)(m0 + (tid >> 1)) * K + kb + GBK +
                             (tid & 1) * 32, 0, 3);
      __builtin_prefetch(B + (size_t)(kb + GBK + (tid >> 5)) * N + n0 +
                             (tid & 31) * 4, 0, 3);
    }

    // ---- compute: two K=32 WMMA steps over the 64-wide stage ----
#pragma unroll
    for (int ks = 0; ks < GBK; ks += 32) {
      const int kf = ks + hb * 8;
      bf16x16 afr[2], bfr[4];
#pragma unroll
      for (int mi = 0; mi < 2; ++mi)
        afr[mi] = frag_from_lds(&sA[wr * 32 + mi * 16 + ln][kf]);
#pragma unroll
      for (int ni = 0; ni < 4; ++ni)
        bfr[ni] = frag_from_lds(&sB[wc * 64 + ni * 16 + ln][kf]);
#pragma unroll
      for (int mi = 0; mi < 2; ++mi)
#pragma unroll
        for (int ni = 0; ni < 4; ++ni)
          acc[mi][ni] = __builtin_amdgcn_wmma_f32_16x16x32_bf16(
              false, afr[mi], false, bfr[ni], (short)0, acc[mi][ni], false, false);
    }
  }

  // ---- epilogue: C/D layout -> global fp32 ----
#pragma unroll
  for (int mi = 0; mi < 2; ++mi)
#pragma unroll
    for (int ni = 0; ni < 4; ++ni)
#pragma unroll
      for (int r = 0; r < 8; ++r) {
        const int row = m0 + wr * 32 + mi * 16 + r + hb * 8;
        const int col = n0 + wc * 64 + ni * 16 + ln;
        if (col < N) C[(size_t)row * N + col] = acc[mi][ni][r];
      }
}

// ---------------------------------------------------------------------------
// RMS norm: one block per row. out[i] = in[i] * rsqrt(mean(in^2)+eps) * w[i]
// ---------------------------------------------------------------------------
__global__ __launch_bounds__(256) void rmsnorm_kernel(
    const float* __restrict__ in, const float* __restrict__ w,
    float* __restrict__ out, int L, int inStride, int outStride) {
  const int row = blockIdx.x;
  const float* x = in + (size_t)row * inStride;
  float* y = out + (size_t)row * outStride;

  float s = 0.0f;
  for (int i = threadIdx.x; i < L; i += 256) { const float v = x[i]; s += v * v; }
#pragma unroll
  for (int m = 1; m < 32; m <<= 1) s += __shfl_xor(s, m, 32);

  __shared__ float red[8];
  __shared__ float scale;
  if ((threadIdx.x & 31) == 0) red[threadIdx.x >> 5] = s;
  __syncthreads();
  if (threadIdx.x == 0) {
    float t = 0.0f;
#pragma unroll
    for (int i = 0; i < 8; ++i) t += red[i];
    scale = rsqrtf(t / (float)L + 1e-6f);
  }
  __syncthreads();
  const float r = scale;
  for (int i = threadIdx.x; i < L; i += 256) y[i] = x[i] * r * w[i];
}

// ---------------------------------------------------------------------------
// RoPE + state assembly (see reference _reorder/_rotate_half analysis):
//   j<32 : rot[j] = r[2j]*cos   - r[2j+1]*sin
//   j>=32: rot[j] = r[2j'+1]*cos + r[2j']*sin   (j' = j-32), freq idx = j&31
// ---------------------------------------------------------------------------
__global__ __launch_bounds__(128) void rope_assemble_kernel(
    const float* __restrict__ t_q, const float* __restrict__ t_kv,
    const float* __restrict__ t_ckv, const int* __restrict__ position,
    float* __restrict__ q_state, float* __restrict__ k_state,
    float* __restrict__ v_buf, int S) {
  const int s = blockIdx.x, h = blockIdx.y, d = threadIdx.x;
  const float* qrow  = t_q  + (size_t)s * (MLA_H * MLA_DQK) + h * MLA_DQK;
  const float* kvrow = t_kv + (size_t)s * (MLA_H * 256) + h * 256;
  float* qst = q_state + ((size_t)h * S + s) * MLA_DQK;
  float* kst = k_state + ((size_t)h * S + s) * MLA_DQK;
  float* vr  = v_buf   + ((size_t)h * S + s) * MLA_DV;

  qst[d] = qrow[d];          // q_nope
  kst[d] = kvrow[d];         // k_nope
  vr[d]  = kvrow[128 + d];   // v

  if (d < 64) {
    const float t = (float)position[s];
    const int jm = d & 31;
    const float invf = __expf(-(float)jm * (1.0f / 32.0f) * 9.210340371976184f);
    const float ang = t * invf;
    const float c = __cosf(ang), sn = __sinf(ang);
    const float* qr = qrow + 128;
    const float* kr = t_ckv + (size_t)s * (MLA_KVLR + MLA_DR) + MLA_KVLR;
    float qv, kv2;
    if (d < 32) {
      qv  = qr[2 * d] * c - qr[2 * d + 1] * sn;
      kv2 = kr[2 * d] * c - kr[2 * d + 1] * sn;
    } else {
      const int j = d - 32;
      qv  = qr[2 * j + 1] * c + qr[2 * j] * sn;
      kv2 = kr[2 * j + 1] * c + kr[2 * j] * sn;
    }
    qst[128 + d] = qv;
    kst[128 + d] = kv2;
  }
}

// ---------------------------------------------------------------------------
// Causal flash attention, one head per blockIdx.y, 128 query rows per block.
// Block = 256 threads (8 waves); wave w owns rows q0 = bx*128 + w*16.
// The block lock-steps over 32-key tiles; K (32x192, B-fragment layout) and
// V (transposed to [d][key]) are staged ONCE per block into LDS as bf16.
// Staging issues all global loads before converting (overlapped latency).
// Scores: 12 WMMAs over d=192; online softmax with 16-lane shuffles; P goes
// C-layout -> A-layout through wave-private LDS ordered by s_wait_dscnt;
// P@V: 8 WMMAs into fp32 accumulators.
// ---------------------------------------------------------------------------
__global__ __launch_bounds__(256) void mla_attention_kernel(
    const float* __restrict__ q_state,  // [H][S][192]
    const float* __restrict__ k_state,  // [H][S][192]
    const float* __restrict__ v_buf,    // [H][S][128]
    float* __restrict__ out,            // [S][H*128]
    int S) {
  __shared__ u16 sK[32][MLA_DQK + 8];  // [key][kdim], 400B rows (16B aligned)
  __shared__ u16 sV[MLA_DV][40];       // [d][key], 80B rows
  __shared__ u16 sP[8][16][32];        // per-wave P tile (A layout)

  const int tid  = threadIdx.x;
  const int h    = blockIdx.y;
  const int lane = tid & 31;
  const int wave = tid >> 5;
  const int hb   = lane >> 4;
  const int ln   = lane & 15;
  const int q0   = blockIdx.x * 128 + wave * 16;
  const float scale = 0.08838834764831845f;  // 1/sqrt(128)

  const float* qbase = q_state + (size_t)h * S * MLA_DQK;
  const float* kbase = k_state + (size_t)h * S * MLA_DQK;
  const float* vbase = v_buf   + (size_t)h * S * MLA_DV;

  // Q fragments (A-matrix layout) for this wave's 16 rows, 6 K=32 chunks.
  bf16x16 qf[6];
  {
    const float* qrow = qbase + (size_t)(q0 + ln) * MLA_DQK;
#pragma unroll
    for (int f = 0; f < 6; ++f) qf[f] = frag_from_global(qrow, f * 32 + hb * 8);
  }

  f32x8 acc[8];
#pragma unroll
  for (int i = 0; i < 8; ++i) acc[i] = (f32x8)0.0f;
  float m_i[8], l_i[8];
#pragma unroll
  for (int r = 0; r < 8; ++r) { m_i[r] = -__builtin_inff(); l_i[r] = 0.0f; }

  const int kend_w = q0 + 16;                      // wave's causal key bound
  int kend_b = (blockIdx.x + 1) * 128;             // block's key bound
  if (kend_b > S) kend_b = S;

  for (int kb = 0; kb < kend_b; kb += 32) {
    __syncthreads();
    // ---- stage K (32x192) and V (32x128, transposed): load all, then cvt ----
    float4 vk[6], vv[4];
#pragma unroll
    for (int i = 0; i < 6; ++i) {
      const int fi  = tid + i * 256;               // 0..1535 float4 units
      const int key = fi / 48;
      const int c4  = fi - key * 48;
      vk[i] = *(const float4*)(kbase + (size_t)(kb + key) * MLA_DQK + c4 * 4);
    }
#pragma unroll
    for (int i = 0; i < 4; ++i) {
      const int fi  = tid + i * 256;               // 0..1023 float4 units
      const int key = fi >> 5;
      const int c4  = fi & 31;
      vv[i] = *(const float4*)(vbase + (size_t)(kb + key) * MLA_DV + c4 * 4);
    }
#pragma unroll
    for (int i = 0; i < 6; ++i) {
      const int fi  = tid + i * 256;
      const int key = fi / 48;
      const int c4  = fi - key * 48;
      const unsigned lo = pk2bf(vk[i].x, vk[i].y);
      const unsigned hi = pk2bf(vk[i].z, vk[i].w);
      *(unsigned long long*)(&sK[key][c4 * 4]) =
          (unsigned long long)lo | ((unsigned long long)hi << 32);
    }
#pragma unroll
    for (int i = 0; i < 4; ++i) {
      const int fi  = tid + i * 256;
      const int key = fi >> 5;
      const int d   = (fi & 31) * 4;
      sV[d + 0][key] = f2bf(vv[i].x);
      sV[d + 1][key] = f2bf(vv[i].y);
      sV[d + 2][key] = f2bf(vv[i].z);
      sV[d + 3][key] = f2bf(vv[i].w);
    }
    __syncthreads();

    if (kb >= kend_w) continue;  // fully-masked for this wave; barriers only

    // ---- scores: two 16x16 C tiles over the 32-key tile ----
    f32x8 s0 = (f32x8)0.0f, s1 = (f32x8)0.0f;
#pragma unroll
    for (int f = 0; f < 6; ++f) {
      const bf16x16 kf0 = frag_from_lds(&sK[ln][f * 32 + hb * 8]);
      const bf16x16 kf1 = frag_from_lds(&sK[16 + ln][f * 32 + hb * 8]);
      s0 = __builtin_amdgcn_wmma_f32_16x16x32_bf16(false, qf[f], false, kf0,
                                                   (short)0, s0, false, false);
      s1 = __builtin_amdgcn_wmma_f32_16x16x32_bf16(false, qf[f], false, kf1,
                                                   (short)0, s1, false, false);
    }

    // ---- online softmax (rows live across 16-lane half-groups) ----
    const int k0i = kb + ln;
    const int k1i = kb + 16 + ln;
    float alpha[8];
#pragma unroll
    for (int r = 0; r < 8; ++r) {
      const int row = q0 + r + hb * 8;
      const float x0 = (k0i <= row) ? s0[r] * scale : -__builtin_inff();
      const float x1 = (k1i <= row) ? s1[r] * scale : -__builtin_inff();
      const float rm = hmax16(fmaxf(x0, x1));
      const float mn = fmaxf(m_i[r], rm);
      const float a  = __expf(m_i[r] - mn);
      const float p0 = __expf(x0 - mn);
      const float p1 = __expf(x1 - mn);
      l_i[r] = l_i[r] * a + hsum16(p0 + p1);
      m_i[r] = mn;
      alpha[r] = a;
      const int m = r + hb * 8;
      sP[wave][m][ln]      = f2bf(p0);
      sP[wave][m][ln + 16] = f2bf(p1);
    }
#pragma unroll
    for (int t = 0; t < 8; ++t)
#pragma unroll
      for (int r = 0; r < 8; ++r) acc[t][r] *= alpha[r];

    // Per-wave LDS RAW: DS ops are in-order within a wave; drain DScnt.
    asm volatile("s_wait_dscnt 0" ::: "memory");

    const bf16x16 pf = frag_from_lds(&sP[wave][ln][hb * 8]);

    // ---- P @ V: 8 output column tiles of 16 ----
#pragma unroll
    for (int nt = 0; nt < 8; ++nt) {
      const bf16x16 vf = frag_from_lds(&sV[nt * 16 + ln][hb * 8]);
      acc[nt] = __builtin_amdgcn_wmma_f32_16x16x32_bf16(
          false, pf, false, vf, (short)0, acc[nt], false, false);
    }
  }

  // ---- epilogue: out[s][h*128 + d] = acc / l ----
#pragma unroll
  for (int r = 0; r < 8; ++r) {
    const float inv_l = 1.0f / l_i[r];
    const int row = q0 + r + hb * 8;
#pragma unroll
    for (int nt = 0; nt < 8; ++nt) {
      const int d = nt * 16 + ln;
      out[(size_t)row * (MLA_H * MLA_DV) + h * MLA_DV + d] = acc[nt][r] * inv_l;
    }
  }
}

// ---------------------------------------------------------------------------
// Host-side orchestration.
// ---------------------------------------------------------------------------
extern "C" void kernel_launch(void* const* d_in, const int* in_sizes, int n_in,
                              void* d_out, int out_size, void* d_ws, size_t ws_size,
                              hipStream_t stream) {
  (void)in_sizes; (void)n_in; (void)out_size; (void)ws_size;

  const float* x        = (const float*)d_in[0];
  const float* Wq_down  = (const float*)d_in[1];
  const float* q_norm_w = (const float*)d_in[2];
  const float* Wq_up    = (const float*)d_in[3];
  const float* Wkv_down = (const float*)d_in[4];
  const float* kv_norm_w= (const float*)d_in[5];
  const float* Wkv_up   = (const float*)d_in[6];
  const float* Wo       = (const float*)d_in[7];
  const int*   position = (const int*)d_in[8];
  // d_in[9] = causal mask (tril) -- implicit in the attention kernel.
  float* out = (float*)d_out;

  const int S = MLA_S;
  float* ws = (float*)d_ws;
  size_t o = 0;
  float* t_qdown = ws + o; o += (size_t)S * MLA_QLR;
  float* t_qn    = ws + o; o += (size_t)S * MLA_QLR;
  float* t_q     = ws + o; o += (size_t)S * (MLA_H * MLA_DQK);
  float* t_ckv   = ws + o; o += (size_t)S * (MLA_KVLR + MLA_DR);
  float* t_ckvn  = ws + o; o += (size_t)S * MLA_KVLR;
  float* t_kv    = ws + o; o += (size_t)S * (MLA_H * (MLA_DN + MLA_DV));
  float* q_state = ws + o; o += (size_t)MLA_H * S * MLA_DQK;
  float* k_state = ws + o; o += (size_t)MLA_H * S * MLA_DQK;
  float* v_buf   = ws + o; o += (size_t)MLA_H * S * MLA_DV;
  float* att_out = ws + o; o += (size_t)S * (MLA_H * MLA_DV);

  const dim3 tb(256);
  auto ggrid = [](int M, int N) { return dim3((N + GBN - 1) / GBN, M / GBM); };

  // 1) q_down = x @ Wq_down  [2048,2048]x[2048,1536]
  gemm_bf16_wmma<<<ggrid(S, MLA_QLR), tb, 0, stream>>>(x, Wq_down, t_qdown,
                                                       S, MLA_QLR, MLA_HID);
  // 2) RMS norm (q)
  rmsnorm_kernel<<<dim3(S), tb, 0, stream>>>(t_qdown, q_norm_w, t_qn,
                                             MLA_QLR, MLA_QLR, MLA_QLR);
  // 3) q = q_norm @ Wq_up  [2048,1536]x[1536,3072]
  gemm_bf16_wmma<<<ggrid(S, MLA_H * MLA_DQK), tb, 0, stream>>>(
      t_qn, Wq_up, t_q, S, MLA_H * MLA_DQK, MLA_QLR);
  // 4) ckv = x @ Wkv_down  [2048,2048]x[2048,576]
  gemm_bf16_wmma<<<ggrid(S, MLA_KVLR + MLA_DR), tb, 0, stream>>>(
      x, Wkv_down, t_ckv, S, MLA_KVLR + MLA_DR, MLA_HID);
  // 5) RMS norm on c (first 512 cols of ckv)
  rmsnorm_kernel<<<dim3(S), tb, 0, stream>>>(t_ckv, kv_norm_w, t_ckvn,
                                             MLA_KVLR, MLA_KVLR + MLA_DR, MLA_KVLR);
  // 6) kv = c @ Wkv_up  [2048,512]x[512,4096]
  gemm_bf16_wmma<<<ggrid(S, MLA_H * (MLA_DN + MLA_DV)), tb, 0, stream>>>(
      t_ckvn, Wkv_up, t_kv, S, MLA_H * (MLA_DN + MLA_DV), MLA_KVLR);
  // 7) RoPE + assemble q_state / k_state / v
  rope_assemble_kernel<<<dim3(S, MLA_H), dim3(128), 0, stream>>>(
      t_q, t_kv, t_ckv, position, q_state, k_state, v_buf, S);
  // 8) causal flash attention
  mla_attention_kernel<<<dim3(S / 128, MLA_H), tb, 0, stream>>>(
      q_state, k_state, v_buf, att_out, S);
  // 9) out = att_out @ Wo  [2048,2048]x[2048,2048]
  gemm_bf16_wmma<<<ggrid(S, MLA_HID), tb, 0, stream>>>(att_out, Wo, out,
                                                       S, MLA_HID, MLA_HID);
}
